// SBERTGCN_24378234372653
// MI455X (gfx1250) — compile-verified, matrix operand order checked by reading, not run on hardware
//
#include <hip/hip_runtime.h>
#include <hip/hip_bf16.h>

typedef __attribute__((ext_vector_type(16))) __bf16 v16bf;
typedef __attribute__((ext_vector_type(8)))  float  v8f;

#define DDIM 128
#define LAMBDA_F 0.7f

__device__ __forceinline__ void atomicAddF(float* p, float v) {
    __hip_atomic_fetch_add(p, v, __ATOMIC_RELAXED, __HIP_MEMORY_SCOPE_AGENT);
}

// ---------------- weight convert: W[K x Dout] fp32 -> WT[Dout x K] bf16 (transposed) ----
__global__ void wt_bf16_kernel(const float* __restrict__ W, __bf16* __restrict__ WT,
                               int K, int Dout) {
    int gid = blockIdx.x * blockDim.x + threadIdx.x;
    if (gid >= K * Dout) return;
    int k = gid / Dout, c = gid % Dout;
    WT[(size_t)c * K + k] = (__bf16)W[gid];
}

// ---------------- GEMM: Y[M x Dout] = X[M x K] @ W  (W given as bf16 WT[Dout x K]) -----
// One wave computes a 16x64 output strip = 4 WMMA tiles; the fp32->bf16 A fragment is
// loaded/converted once per K-step and reused across 4 v_wmma_f32_16x16x32_bf16 issues
// (independent accumulators also cover the WMMA->WMMA RAW hazard window).
// Requires M%16==0, K%32==0, Dout%64==0. Whole-wave early exit keeps EXEC all-ones.
__global__ void gemm_bf16_wmma_kernel(const float* __restrict__ X,
                                      const __bf16* __restrict__ WT,
                                      float* __restrict__ Y,
                                      int M, int K, int Dout) {
    const int lane    = threadIdx.x & 31;
    const int wave    = (blockIdx.x * blockDim.x + threadIdx.x) >> 5;
    const int stripsN = Dout >> 6;             // 64-wide column strips
    const int total   = (M >> 4) * stripsN;
    if (wave >= total) return;                 // whole-wave exit, EXEC stays full
    const int mt = wave / stripsN;
    const int ns = wave % stripsN;
    const int hf = lane >> 4;                  // K-half selector
    const int ml = lane & 15;                  // row (A) / col (B) within tile

    const float*  xrow = X  + (size_t)(mt * 16 + ml) * K;
    const __bf16* wb0  = WT + (size_t)(ns * 64 + ml) * K;   // col tile 0; tile t at +t*16*K

    v8f acc0 = {}, acc1 = {}, acc2 = {}, acc3 = {};
    for (int kk = 0; kk < K; kk += 32) {
        v16bf a, b0, b1, b2, b3;
#pragma unroll
        for (int j = 0; j < 8; ++j) {
            // 16-bit A/B fragment layout: VGPR j <-> K = 16*(j>=4) + 8*half + 2*(j&3)
            int k0 = kk + ((j & 4) << 2) + (hf << 3) + ((j & 3) << 1);
            float a0 = xrow[k0], a1 = xrow[k0 + 1];
            a[2 * j]      = (__bf16)a0;
            a[2 * j + 1]  = (__bf16)a1;
            b0[2 * j]     = wb0[k0];
            b0[2 * j + 1] = wb0[k0 + 1];
            b1[2 * j]     = wb0[(size_t)16 * K + k0];
            b1[2 * j + 1] = wb0[(size_t)16 * K + k0 + 1];
            b2[2 * j]     = wb0[(size_t)32 * K + k0];
            b2[2 * j + 1] = wb0[(size_t)32 * K + k0 + 1];
            b3[2 * j]     = wb0[(size_t)48 * K + k0];
            b3[2 * j + 1] = wb0[(size_t)48 * K + k0 + 1];
        }
        acc0 = __builtin_amdgcn_wmma_f32_16x16x32_bf16(false, a, false, b0, (short)0, acc0, false, false);
        acc1 = __builtin_amdgcn_wmma_f32_16x16x32_bf16(false, a, false, b1, (short)0, acc1, false, false);
        acc2 = __builtin_amdgcn_wmma_f32_16x16x32_bf16(false, a, false, b2, (short)0, acc2, false, false);
        acc3 = __builtin_amdgcn_wmma_f32_16x16x32_bf16(false, a, false, b3, (short)0, acc3, false, false);
    }

    float* yrowbase = Y + (size_t)ns * 64 + ml;
#pragma unroll
    for (int r = 0; r < 8; ++r) {
        int row = mt * 16 + r + hf * 8;        // C/D layout: VGPR r -> M = r + 8*half
        float* yr = yrowbase + (size_t)row * Dout;
        yr[0]  = acc0[r];
        yr[16] = acc1[r];
        yr[32] = acc2[r];
        yr[48] = acc3[r];
    }
}

// ---------------- scatter rows: Y[idx[i], :] = T[i, :] ---------------------------------
__global__ void row_set_kernel(float* __restrict__ Y, const int* __restrict__ idx,
                               const float* __restrict__ T, int B) {
    int gid = blockIdx.x * blockDim.x + threadIdx.x;
    if (gid >= B * DDIM) return;
    int i = gid >> 7, d = gid & 127;
    Y[(size_t)idx[i] * DDIM + d] = T[gid];
}

// ---------------- degree: init to 1 (self-loop), scatter edge weights, rsqrt -----------
__global__ void deg_init_kernel(float* __restrict__ deg, int N) {
    int i = blockIdx.x * blockDim.x + threadIdx.x;
    if (i < N) deg[i] = 1.0f;
}
__global__ void deg_scatter_kernel(float* __restrict__ deg, const int* __restrict__ dst,
                                   const float* __restrict__ ew, int E) {
    int e = blockIdx.x * blockDim.x + threadIdx.x;
    if (e < E) atomicAddF(&deg[dst[e]], ew[e]);
}
__global__ void dinv_kernel(float* __restrict__ deg, int N) {
    int i = blockIdx.x * blockDim.x + threadIdx.x;
    if (i >= N) return;
    float v = deg[i];
    deg[i] = (v > 0.0f) ? __frsqrt_rn(v) : 0.0f;
}

// ---------------- self-loop term doubles as accumulator init: OUT = H * dinv^2 --------
__global__ void selfloop_init_kernel(const float* __restrict__ H, float* __restrict__ OUT,
                                     const float* __restrict__ dinv, int N) {
    int gid = blockIdx.x * blockDim.x + threadIdx.x;
    if (gid >= N * DDIM) return;
    float s = dinv[gid >> 7];
    OUT[gid] = H[gid] * (s * s);
}

// ---------------- edge message passing: one wave per edge, 4 dims/lane -----------------
__global__ void edge_scatter_kernel(const float* __restrict__ H, float* __restrict__ OUT,
                                    const int* __restrict__ src, const int* __restrict__ dst,
                                    const float* __restrict__ ew,
                                    const float* __restrict__ dinv, int E) {
    int e = (blockIdx.x * blockDim.x + threadIdx.x) >> 5;
    if (e >= E) return;
    int lane = threadIdx.x & 31;
    int s = src[e], d = dst[e];
    float coef = dinv[s] * ew[e] * dinv[d];
    const float4 v = *(const float4*)(H + (size_t)s * DDIM + lane * 4);
    float* o = OUT + (size_t)d * DDIM + lane * 4;
    atomicAddF(o + 0, v.x * coef);
    atomicAddF(o + 1, v.y * coef);
    atomicAddF(o + 2, v.z * coef);
    atomicAddF(o + 3, v.w * coef);
}

__global__ void relu_kernel(float* __restrict__ A, int n) {
    int gid = blockIdx.x * blockDim.x + threadIdx.x;
    if (gid < n) A[gid] = fmaxf(A[gid], 0.0f);
}

// ---------------- outputs: c = LAMBDA*sp + (1-LAMBDA)*xg[idx] --------------------------
__global__ void output_kernel(const float* __restrict__ s1p, const float* __restrict__ s2p,
                              const float* __restrict__ xg,
                              const int* __restrict__ idx1, const int* __restrict__ idx2,
                              float* __restrict__ out, int B) {
    int gid = blockIdx.x * blockDim.x + threadIdx.x;
    if (gid >= B * DDIM) return;
    int i = gid >> 7, d = gid & 127;
    float g1 = xg[(size_t)idx1[i] * DDIM + d];
    float g2 = xg[(size_t)idx2[i] * DDIM + d];
    out[gid]            = LAMBDA_F * s1p[gid] + (1.0f - LAMBDA_F) * g1;
    out[B * DDIM + gid] = LAMBDA_F * s2p[gid] + (1.0f - LAMBDA_F) * g2;
}

extern "C" void kernel_launch(void* const* d_in, const int* in_sizes, int n_in,
                              void* d_out, int out_size, void* d_ws, size_t ws_size,
                              hipStream_t stream) {
    const float* node_emb = (const float*)d_in[0];
    const float* s1       = (const float*)d_in[1];
    const float* s2       = (const float*)d_in[2];
    const float* w_proj   = (const float*)d_in[3];
    const float* w1       = (const float*)d_in[4];
    const float* w2       = (const float*)d_in[5];
    const int*   idx1     = (const int*)d_in[6];
    const int*   idx2     = (const int*)d_in[7];
    const int*   eidx     = (const int*)d_in[8];
    const float* ew       = (const float*)d_in[9];

    const int N  = in_sizes[0] / DDIM;   // 100000
    const int B  = in_sizes[6];          // 1024
    const int E  = in_sizes[9];          // 1.6M
    const int SD = in_sizes[1] / B;      // 384
    const int* esrc = eidx;
    const int* edst = eidx + E;

    // ---- workspace carve (256B aligned) ----
    char* p = (char*)d_ws;
    auto carve = [&](size_t bytes) -> char* {
        char* r = p; p += (bytes + 255) & ~(size_t)255; return r;
    };
    float*  s1p  = (float*)carve((size_t)B * DDIM * 4);
    float*  s2p  = (float*)carve((size_t)B * DDIM * 4);
    float*  t1   = (float*)carve((size_t)B * DDIM * 4);
    float*  t2   = (float*)carve((size_t)B * DDIM * 4);
    __bf16* wpT  = (__bf16*)carve((size_t)SD * DDIM * 2);
    __bf16* w1T  = (__bf16*)carve((size_t)DDIM * DDIM * 2);
    __bf16* w2T  = (__bf16*)carve((size_t)DDIM * DDIM * 2);
    float*  dinv = (float*)carve((size_t)N * 4);
    float*  bufA = (float*)carve((size_t)N * DDIM * 4);   // h1, later h2
    float*  bufB = (float*)carve((size_t)N * DDIM * 4);   // agg1/h, later x_g

    const int T = 256;
    auto cdiv = [](long long a, long long b) { return (int)((a + b - 1) / b); };
    auto gemm = [&](const float* X, const __bf16* WT, float* Y, int M, int K) {
        int waves = (M >> 4) * (DDIM >> 6);   // one wave per 16x64 strip
        hipLaunchKernelGGL(gemm_bf16_wmma_kernel, dim3(cdiv(waves, 8)), dim3(T), 0, stream,
                           X, WT, Y, M, K, DDIM);
    };

    // weights -> transposed bf16
    hipLaunchKernelGGL(wt_bf16_kernel, dim3(cdiv((long long)SD * DDIM, T)), dim3(T), 0, stream, w_proj, wpT, SD, DDIM);
    hipLaunchKernelGGL(wt_bf16_kernel, dim3(cdiv(DDIM * DDIM, T)), dim3(T), 0, stream, w1, w1T, DDIM, DDIM);
    hipLaunchKernelGGL(wt_bf16_kernel, dim3(cdiv(DDIM * DDIM, T)), dim3(T), 0, stream, w2, w2T, DDIM, DDIM);

    // sbert projections (WMMA)
    gemm(s1, wpT, s1p, B, SD);
    gemm(s2, wpT, s2p, B, SD);

    // h1 = node_emb @ w1, then patch rows idx1 -> s1p@w1, idx2 -> s2p@w1
    gemm(node_emb, w1T, bufA, N, DDIM);
    gemm(s1p, w1T, t1, B, DDIM);
    gemm(s2p, w1T, t2, B, DDIM);
    hipLaunchKernelGGL(row_set_kernel, dim3(cdiv(B * DDIM, T)), dim3(T), 0, stream, bufA, idx1, t1, B);
    hipLaunchKernelGGL(row_set_kernel, dim3(cdiv(B * DDIM, T)), dim3(T), 0, stream, bufA, idx2, t2, B);

    // symmetric normalization coefficients
    hipLaunchKernelGGL(deg_init_kernel, dim3(cdiv(N, T)), dim3(T), 0, stream, dinv, N);
    hipLaunchKernelGGL(deg_scatter_kernel, dim3(cdiv(E, T)), dim3(T), 0, stream, dinv, edst, ew, E);
    hipLaunchKernelGGL(dinv_kernel, dim3(cdiv(N, T)), dim3(T), 0, stream, dinv, N);

    // layer 1: self-loop init + edge scatter + relu
    hipLaunchKernelGGL(selfloop_init_kernel, dim3(cdiv((long long)N * DDIM, T)), dim3(T), 0, stream, bufA, bufB, dinv, N);
    hipLaunchKernelGGL(edge_scatter_kernel, dim3(cdiv(E, 8)), dim3(T), 0, stream, bufA, bufB, esrc, edst, ew, dinv, E);
    hipLaunchKernelGGL(relu_kernel, dim3(cdiv((long long)N * DDIM, T)), dim3(T), 0, stream, bufB, N * DDIM);

    // layer 2: h2 = h @ w2 (WMMA), then self-loop init + edge scatter -> x_g in bufB
    gemm(bufB, w2T, bufA, N, DDIM);
    hipLaunchKernelGGL(selfloop_init_kernel, dim3(cdiv((long long)N * DDIM, T)), dim3(T), 0, stream, bufA, bufB, dinv, N);
    hipLaunchKernelGGL(edge_scatter_kernel, dim3(cdiv(E, 8)), dim3(T), 0, stream, bufA, bufB, esrc, edst, ew, dinv, E);

    // outputs
    hipLaunchKernelGGL(output_kernel, dim3(cdiv(B * DDIM, T)), dim3(T), 0, stream,
                       s1p, s2p, bufB, idx1, idx2, (float*)d_out, B);
}
